// TxGNNModel_67877663146323
// MI455X (gfx1250) — matrix operand order repaired and checked by reading, not compiled, not run on hardware
//
#include <hip/hip_runtime.h>

// ---------------------------------------------------------------------------
// Types for CDNA5 WMMA (wave32): bf16 16x16x32, fp32 accumulate.
// ---------------------------------------------------------------------------
typedef __attribute__((ext_vector_type(16))) __bf16 v16bf;
typedef __attribute__((ext_vector_type(8)))  __bf16 v8bf;
typedef __attribute__((ext_vector_type(4)))  __bf16 v4bf;
typedef __attribute__((ext_vector_type(8)))  float  v8f;

#define DD 128   // hidden dim
#define RR 8     // relations
#define KSTK (DD * (RR + 1))   // 1152 = stacked K for RGCN layer GEMM

__device__ __forceinline__ __bf16 f2bf(float f) {
  unsigned u = __builtin_bit_cast(unsigned, f);
  unsigned r = (u + 0x7FFFu + ((u >> 16) & 1u)) >> 16;   // round-to-nearest-even
  unsigned short h = (unsigned short)r;
  return __builtin_bit_cast(__bf16, h);
}
__device__ __forceinline__ float bf2f(__bf16 b) {
  unsigned short h = __builtin_bit_cast(unsigned short, b);
  unsigned u = ((unsigned)h) << 16;
  return __builtin_bit_cast(float, u);
}

// ---------------------------------------------------------------------------
// Edge counting: cnt[dst*R + type] += 1, deg[dst] += 1  (fp32 atomics)
// ---------------------------------------------------------------------------
__global__ void count_kernel(const int* __restrict__ dst, const int* __restrict__ et,
                             float* __restrict__ cnt, float* __restrict__ deg, int E) {
  int e = blockIdx.x * blockDim.x + threadIdx.x;
  if (e >= E) return;
  int d = dst[e];
  atomicAdd(&cnt[d * RR + et[e]], 1.0f);
  atomicAdd(&deg[d], 1.0f);
}

// ---------------------------------------------------------------------------
// fp32 -> bf16 array conversion
// ---------------------------------------------------------------------------
__global__ void cvt_bf16(const float* __restrict__ in, __bf16* __restrict__ out, size_t n) {
  size_t i = (size_t)blockIdx.x * blockDim.x + threadIdx.x;
  if (i < n) out[i] = f2bf(in[i]);
}

// ---------------------------------------------------------------------------
// Concat [W_self ; W_rel(0..7)] -> fp32 [1152 x 128] row-major
// ---------------------------------------------------------------------------
__global__ void concat_w(const float* __restrict__ wself, const float* __restrict__ wrel,
                         float* __restrict__ wstack) {
  int t = blockIdx.x * blockDim.x + threadIdx.x;
  if (t >= KSTK * DD) return;
  int col = t & (DD - 1);
  int row = t >> 7;
  wstack[t] = (row < DD) ? wself[row * DD + col]
                         : wrel[(size_t)(row - DD) * DD + col];
}

// ---------------------------------------------------------------------------
// Pack a fp32 [K x 128] matrix (optionally row-gathered) into the per-lane
// WMMA B-fragment layout for v_wmma_f32_16x16x32_bf16:
//   element t = ((kt*8 + jt)*32 + lane)*16 + e
//   maps to  K = kt*32 + 16*(lane>>4) + e,  n = jt*16 + (lane&15)
// GEMM then reads B with two coalesced 16B loads per lane per K-tile.
// ---------------------------------------------------------------------------
__global__ void pack_b(const float* __restrict__ src, const int* __restrict__ rowidx,
                       __bf16* __restrict__ Bp, int K) {
  size_t t = (size_t)blockIdx.x * blockDim.x + threadIdx.x;
  if (t >= (size_t)K * DD) return;
  int e      = (int)(t & 15);
  size_t f   = t >> 4;
  int lane   = (int)(f & 31);
  size_t g   = f >> 5;
  int jt     = (int)(g & 7);
  int kt     = (int)(g >> 3);
  int k = kt * 32 + 16 * (lane >> 4) + e;
  int n = jt * 16 + (lane & 15);
  int row = rowidx ? rowidx[k] : k;
  Bp[t] = f2bf(src[(size_t)row * DD + n]);
}

// ---------------------------------------------------------------------------
// Message scatter: one wave per edge; lane handles 4 of the 128 features.
// msg[(dst*R + type)*128 + c] += h[src*128 + c]   (fp32 atomics, bf16 source)
// ---------------------------------------------------------------------------
__global__ void scatter_msgs(const __bf16* __restrict__ h, const int* __restrict__ src,
                             const int* __restrict__ dst, const int* __restrict__ et,
                             float* __restrict__ msg, int E) {
  size_t gid = (size_t)blockIdx.x * blockDim.x + threadIdx.x;
  size_t e = gid >> 5;
  if (e >= (size_t)E) return;
  int l = (int)(gid & 31);
  const __bf16* hp = h + (size_t)src[e] * DD + l * 4;
  float* mp = msg + ((size_t)dst[e] * RR + et[e]) * DD + l * 4;
  v4bf hv = *(const v4bf*)hp;
  atomicAdd(mp + 0, bf2f(hv[0]));
  atomicAdd(mp + 1, bf2f(hv[1]));
  atomicAdd(mp + 2, bf2f(hv[2]));
  atomicAdd(mp + 3, bf2f(hv[3]));
}

// ---------------------------------------------------------------------------
// Build the stacked A matrix [N x 1152] bf16: [ h_self | msg/max(cnt,1) ]
// ---------------------------------------------------------------------------
__global__ void build_A(const __bf16* __restrict__ hself, const float* __restrict__ msg,
                        const float* __restrict__ cnt, __bf16* __restrict__ A, int Nn) {
  size_t i = (size_t)blockIdx.x * blockDim.x + threadIdx.x;
  if (i >= (size_t)Nn * KSTK) return;
  int col = (int)(i % KSTK);
  int n   = (int)(i / KSTK);
  __bf16 v;
  if (col < DD) {
    v = hself[(size_t)n * DD + col];
  } else {
    int rc  = col - DD;                 // r*128 + k
    int seg = n * RR + (rc >> 7);
    float inv = 1.0f / fmaxf(cnt[seg], 1.0f);
    v = f2bf(msg[(size_t)seg * DD + (rc & (DD - 1))] * inv);
  }
  A[i] = v;
}

// ---------------------------------------------------------------------------
// WMMA GEMM:  out[M x 128] = A[M x K] * B[K x 128]   (bf16 in, fp32 acc)
// Block = 256 threads = 8 waves; block covers (16*UT) rows x 128 cols.
// Wave w owns column tile jt=w and UT row subtiles -> UT independent
// accumulator chains sharing one B fragment per 32-wide K tile:
//   per K-tile: 2 B loads + 2*UT A loads for UT WMMAs  (UT=4 -> 2.5 ld/wmma)
// All addresses are rebuilt from the A/Bpack kernargs via scalar element
// offsets each iteration (no pointer phis / pointer arrays) so the
// infer-address-space pass keeps every access on the GLOBAL path
// (global_load_b128, not flat_load_b128 which would also tick DScnt).
// A-fragment: lane L (m=L&15, s=L>>4) reads the contiguous 16-half run at
// K-tile offset 16s (matches pack_b's B k order).
// B read from pre-packed fragments (pack_b). Requires M % (16*UT) == 0.
// ---------------------------------------------------------------------------
template <int UT>
__global__ __launch_bounds__(256)
void wmma_gemm_n128(const __bf16* __restrict__ A, const __bf16* __restrict__ Bpack,
                    float* __restrict__ outF, __bf16* __restrict__ outB,
                    int M, int K, int relu) {
  const int tid  = threadIdx.x;
  const int lane = tid & 31;
  const int wv   = tid >> 5;           // 0..7 -> column tile
  const int mt   = blockIdx.x;         // (16*UT)-row stripe
  const int s    = lane >> 4;
  const int kTiles = K >> 5;

  v8f acc[UT];
#pragma unroll
  for (int u = 0; u < UT; ++u)
    acc[u] = (v8f){0.f, 0.f, 0.f, 0.f, 0.f, 0.f, 0.f, 0.f};

  const size_t rowStride = (size_t)16 * K;          // halves between row subtiles
  size_t abase = ((size_t)(mt * 16 * UT + (lane & 15))) * K + 16 * s;
  size_t boff  = ((size_t)wv * 32 + lane) * 16;
  const size_t bstep = 8u * 32u * 16u;              // packed halves per K-tile

  for (int kt = 0; kt < kTiles; ++kt) {
    // B fragment (shared by all UT row subtiles this iteration)
    const __bf16* bp = Bpack + boff;
    v8bf b_lo = *(const v8bf*)(bp);
    v8bf b_hi = *(const v8bf*)(bp + 8);
    __builtin_prefetch(bp + bstep, 0, 3);           // speculative: safe past end
    v16bf b;
#pragma unroll
    for (int i = 0; i < 8; ++i) { b[i] = b_lo[i]; b[i + 8] = b_hi[i]; }

#pragma unroll
    for (int u = 0; u < UT; ++u) {
      const __bf16* ap = A + (abase + (size_t)u * rowStride);
      v8bf a_lo = *(const v8bf*)(ap);
      v8bf a_hi = *(const v8bf*)(ap + 8);
      __builtin_prefetch(ap + 32, 0, 3);
      v16bf a;
#pragma unroll
      for (int i = 0; i < 8; ++i) { a[i] = a_lo[i]; a[i + 8] = a_hi[i]; }
      // D = A*B + C : v_wmma_f32_16x16x32_bf16 (independent chain per u)
      acc[u] = __builtin_amdgcn_wmma_f32_16x16x32_bf16(false, a, false, b,
                                                       (short)0, acc[u], false, false);
    }
    abase += 32;
    boff  += bstep;
  }

  // C/D layout: lane holds col (lane&15), rows 8*(lane>>4)+q for q=0..7
  const int col = wv * 16 + (lane & 15);
#pragma unroll
  for (int u = 0; u < UT; ++u) {
#pragma unroll
    for (int q = 0; q < 8; ++q) {
      int row = mt * 16 * UT + u * 16 + 8 * s + q;
      float v = acc[u][q];
      if (relu) v = v > 0.f ? v : 0.f;
      if (outF) outF[(size_t)row * DD + col] = v;
      if (outB) outB[(size_t)row * DD + col] = f2bf(v);
    }
  }
}

// ---------------------------------------------------------------------------
// Sim row sums: rowsum[i] = sum_j sim[ridx[i], ridx[j]]   (block per row)
// ---------------------------------------------------------------------------
__global__ void sim_rowsum(const float* __restrict__ sim, const int* __restrict__ ridx,
                           float* __restrict__ rowsum, int nd) {
  __shared__ float red[256];
  int i = blockIdx.x;
  const float* rp = sim + (size_t)ridx[i] * nd;
  float sum = 0.f;
  for (int j = threadIdx.x; j < nd; j += 256) sum += rp[ridx[j]];
  red[threadIdx.x] = sum;
  __syncthreads();
  for (int st = 128; st > 0; st >>= 1) {
    if (threadIdx.x < st) red[threadIdx.x] += red[threadIdx.x + st];
    __syncthreads();
  }
  if (threadIdx.x == 0) rowsum[i] = red[0];
}

// ---------------------------------------------------------------------------
// A_sim[i,j] = sim[ridx[i], ridx[j]] / (rowsum[i] + 1e-9)  (bf16, row-major)
// ---------------------------------------------------------------------------
__global__ void build_Asim(const float* __restrict__ sim, const int* __restrict__ ridx,
                           const float* __restrict__ rowsum, __bf16* __restrict__ A, int nd) {
  size_t t = (size_t)blockIdx.x * blockDim.x + threadIdx.x;
  if (t >= (size_t)nd * nd) return;
  int j = (int)(t % nd);
  int i = (int)(t / nd);
  float v = sim[(size_t)ridx[i] * nd + ridx[j]] / (rowsum[i] + 1e-9f);
  A[t] = f2bf(v);
}

// ---------------------------------------------------------------------------
// Blend: out[loc[i], :] = c*h_sim[i,:] + (1-c)*out[loc[i], :]
//        c = 0.7*exp(-0.7*deg[loc[i]]) + 0.2
// ---------------------------------------------------------------------------
__global__ void blend_kernel(const float* __restrict__ h_sim, const int* __restrict__ loc,
                             const float* __restrict__ deg, float* __restrict__ out, int nd) {
  size_t t = (size_t)blockIdx.x * blockDim.x + threadIdx.x;
  if (t >= (size_t)nd * DD) return;
  int j = (int)(t & (DD - 1));
  int i = (int)(t >> 7);
  int g = loc[i];
  float c = 0.7f * __expf(-0.7f * deg[g]) + 0.2f;
  size_t o = (size_t)g * DD + j;
  out[o] = c * h_sim[(size_t)i * DD + j] + (1.0f - c) * out[o];
}

// ---------------------------------------------------------------------------
// Host orchestration
// ---------------------------------------------------------------------------
static inline unsigned grid1(size_t n, int bs) { return (unsigned)((n + bs - 1) / bs); }

extern "C" void kernel_launch(void* const* d_in, const int* in_sizes, int n_in,
                              void* d_out, int out_size, void* d_ws, size_t ws_size,
                              hipStream_t stream) {
  const float* x    = (const float*)d_in[0];
  const float* sim  = (const float*)d_in[1];
  const float* W1s  = (const float*)d_in[2];
  const float* W1r  = (const float*)d_in[3];
  const float* W2s  = (const float*)d_in[4];
  const float* W2r  = (const float*)d_in[5];
  const int* esrc   = (const int*)d_in[6];
  const int* edst   = (const int*)d_in[7];
  const int* etyp   = (const int*)d_in[8];
  const int* bloc   = (const int*)d_in[9];
  const int* mrow   = (const int*)d_in[10];

  const int E  = in_sizes[6];
  const int N  = in_sizes[0] / DD;
  const int ND = in_sizes[9];
  float* out = (float*)d_out;
  (void)esrc; (void)out_size; (void)n_in; (void)ws_size;

  // ---- workspace carve-up (256B aligned) ----
  char* ws = (char*)d_ws;
  size_t cur = 0;
  auto carve = [&](size_t bytes) -> void* {
    void* p = ws + cur;
    cur = (cur + bytes + 255) & ~(size_t)255;
    return p;
  };
  float*  msg    = (float*)carve((size_t)N * RR * DD * sizeof(float));   // 164 MB
  float*  cnt    = (float*)carve((size_t)N * RR * sizeof(float));
  float*  deg    = (float*)carve((size_t)N * sizeof(float));
  __bf16* xb     = (__bf16*)carve((size_t)N * DD * sizeof(__bf16));
  __bf16* h1b    = (__bf16*)carve((size_t)N * DD * sizeof(__bf16));
  __bf16* Abf    = (__bf16*)carve((size_t)N * KSTK * sizeof(__bf16));    // 92 MB
  float*  wstkF  = (float*)carve((size_t)KSTK * DD * sizeof(float));
  __bf16* wpk1   = (__bf16*)carve((size_t)KSTK * DD * sizeof(__bf16));
  __bf16* wpk2   = (__bf16*)carve((size_t)KSTK * DD * sizeof(__bf16));
  float*  rowsum = (float*)carve((size_t)ND * sizeof(float));
  // aliases for the post-RGCN phase (msg/Abf are dead by then)
  __bf16* AsimBf = (__bf16*)msg;                                // ND*ND bf16 (32 MB)
  __bf16* hbPack = (__bf16*)Abf;                                // ND*128 bf16 (1 MB)
  float*  h_sim  = (float*)((char*)Abf + (4u << 20));           // ND*128 fp32 (2 MB)

  const int BS = 256;

  // ---- zero accumulators ----
  hipMemsetAsync(msg, 0, (size_t)N * RR * DD * sizeof(float), stream);
  hipMemsetAsync(cnt, 0, (size_t)N * RR * sizeof(float), stream);
  hipMemsetAsync(deg, 0, (size_t)N * sizeof(float), stream);

  // ---- edge counts / degrees ----
  count_kernel<<<grid1(E, BS), BS, 0, stream>>>(edst, etyp, cnt, deg, E);

  // ---- prepare bf16 operands ----
  cvt_bf16<<<grid1((size_t)N * DD, BS), BS, 0, stream>>>(x, xb, (size_t)N * DD);
  concat_w<<<grid1((size_t)KSTK * DD, BS), BS, 0, stream>>>(W1s, W1r, wstkF);
  pack_b<<<grid1((size_t)KSTK * DD, BS), BS, 0, stream>>>(wstkF, nullptr, wpk1, KSTK);
  concat_w<<<grid1((size_t)KSTK * DD, BS), BS, 0, stream>>>(W2s, W2r, wstkF);
  pack_b<<<grid1((size_t)KSTK * DD, BS), BS, 0, stream>>>(wstkF, nullptr, wpk2, KSTK);

  // ---- layer 1: scatter, build stacked A, WMMA GEMM (+relu) -> h1 (bf16) ----
  scatter_msgs<<<grid1((size_t)E * 32, BS), BS, 0, stream>>>(xb, esrc, edst, etyp, msg, E);
  build_A<<<grid1((size_t)N * KSTK, BS), BS, 0, stream>>>(xb, msg, cnt, Abf, N);
  wmma_gemm_n128<4><<<N / 64, BS, 0, stream>>>(Abf, wpk1, nullptr, h1b, N, KSTK, 1);

  // ---- layer 2: re-zero msg, scatter h1, build A, WMMA GEMM -> out (fp32) ----
  hipMemsetAsync(msg, 0, (size_t)N * RR * DD * sizeof(float), stream);
  scatter_msgs<<<grid1((size_t)E * 32, BS), BS, 0, stream>>>(h1b, esrc, edst, etyp, msg, E);
  build_A<<<grid1((size_t)N * KSTK, BS), BS, 0, stream>>>(h1b, msg, cnt, Abf, N);
  wmma_gemm_n128<4><<<N / 64, BS, 0, stream>>>(Abf, wpk2, out, nullptr, N, KSTK, 0);

  // ---- similarity diffusion ----
  sim_rowsum<<<ND, BS, 0, stream>>>(sim, mrow, rowsum, ND);
  build_Asim<<<grid1((size_t)ND * ND, BS), BS, 0, stream>>>(sim, mrow, rowsum, AsimBf, ND);
  pack_b<<<grid1((size_t)ND * DD, BS), BS, 0, stream>>>(out, bloc, hbPack, ND);
  wmma_gemm_n128<2><<<ND / 32, BS, 0, stream>>>(AsimBf, hbPack, h_sim, nullptr, ND, ND, 0);
  blend_kernel<<<grid1((size_t)ND * DD, BS), BS, 0, stream>>>(h_sim, bloc, deg, out, ND);
}
